// CATDecoderAttention_11458972746231
// MI455X (gfx1250) — compile-verified, matrix operand order checked by reading, not run on hardware
//
#include <hip/hip_runtime.h>
#include <hip/hip_bf16.h>

// ---------------- CDNA5 types ----------------
typedef __bf16 bf16_t;
typedef __attribute__((ext_vector_type(16))) __bf16 v16bf;
typedef __attribute__((ext_vector_type(8)))  __bf16 v8bf;
typedef __attribute__((ext_vector_type(8)))  float  v8f;
typedef __attribute__((ext_vector_type(4)))  unsigned int u32x4;
typedef __attribute__((ext_vector_type(8)))  int i32x8;
typedef __attribute__((ext_vector_type(4)))  int i32x4;

#define T_SEQ 4096
#define HID   1024
#define NH    16
#define NKV   4
#define HD    64
#define ATTN_SCALE 0.125f   // 1/sqrt(64)
#define RMS_EPS 1e-6f
#define NEG_INF (-3.0e38f)

// Load a 16x32 bf16 tile in WMMA 16-bit A/B layout.
// Lanes 0-15: row = lane,    K chunks {0..7} and {16..23}
// Lanes 16-31: row = lane-16, K chunks {8..15} and {24..31}
__device__ __forceinline__ v16bf load_tile16x32(const bf16_t* base, int stride) {
  const int lane = threadIdx.x & 31;
  const int row  = lane & 15;
  const int kb   = (lane & 16) ? 8 : 0;
  const bf16_t* p = base + (size_t)row * stride + kb;
  v8bf lo = *(const v8bf*)(p);        // 16B contiguous -> b128 (global or ds)
  v8bf hi = *(const v8bf*)(p + 16);
  v16bf r;
#pragma unroll
  for (int i = 0; i < 8; ++i) { r[i] = lo[i]; r[8 + i] = hi[i]; }
  return r;
}

__device__ __forceinline__ float redmax16(float v) {
#pragma unroll
  for (int m = 1; m < 16; m <<= 1) v = fmaxf(v, __shfl_xor(v, m, 32));
  return v;
}
__device__ __forceinline__ float redsum16(float v) {
#pragma unroll
  for (int m = 1; m < 16; m <<= 1) v += __shfl_xor(v, m, 32);
  return v;
}

// ---- Tensor Data Mover: 2-D tile load (global -> LDS), bf16 elements ----------
// Descriptor layout per CDNA5 ISA ch. 8.3/8.4 (D# groups 0/1; groups 2/3 unused
// for 2-D tiles). 6-arg builtin form on this toolchain. Args must be wave-uniform.
__device__ __forceinline__ void tdm_load_2d_bf16(
    unsigned lds_addr_bytes, const void* gptr,
    unsigned tile_w, unsigned tile_h,
    unsigned tensor_w, unsigned tensor_h, unsigned row_stride_elems) {
  unsigned long long ga = (unsigned long long)(uintptr_t)gptr;
  u32x4 g0;
  g0[0] = 1u;                                            // count=1, user-mode D#
  g0[1] = lds_addr_bytes;                                // lds_addr
  g0[2] = (unsigned)(ga & 0xFFFFFFFFu);                  // global_addr[31:0]
  g0[3] = (unsigned)((ga >> 32) & 0x1FFFFFFu)            // global_addr[56:32]
        | (2u << 30);                                    // type=2 ("image")
  i32x8 g1;
  g1[0] = (int)(1u << 16);                               // data_size=1 -> 2 bytes
  g1[1] = (int)((tensor_w & 0xFFFFu) << 16);             // tensor_dim0[15:0]
  g1[2] = (int)(((tensor_h & 0xFFFFu) << 16) |           // tensor_dim1[15:0]
                ((tensor_w >> 16) & 0xFFFFu));           // tensor_dim0[31:16]
  g1[3] = (int)(((tile_w & 0xFFFFu) << 16) |             // tile_dim0
                ((tensor_h >> 16) & 0xFFFFu));           // tensor_dim1[31:16]
  g1[4] = (int)(tile_h & 0xFFFFu);                       // tile_dim1 (tile_dim2=0)
  g1[5] = (int)row_stride_elems;                         // tensor_dim0_stride[31:0]
  g1[6] = 0;                                             // stride hi / dim1_stride lo
  g1[7] = 0;
  i32x4 z4 = {0, 0, 0, 0};
  i32x8 z8 = {0, 0, 0, 0, 0, 0, 0, 0};
  __builtin_amdgcn_tensor_load_to_lds(g0, g1, z4, z4, z8, 0);
}

// ---------------- conversion kernels ----------------
__global__ void cvt_f32_bf16(const float* __restrict__ s, bf16_t* __restrict__ d, int n) {
  int i = blockIdx.x * blockDim.x + threadIdx.x;
  if (i < n) d[i] = (bf16_t)s[i];
}

// W[K][N] (row-major) -> Wt[N][K] bf16, so GEMM B-tiles are K-contiguous.
__global__ void transpose_cvt(const float* __restrict__ W, bf16_t* __restrict__ Wt, int K, int N) {
  int i = blockIdx.x * blockDim.x + threadIdx.x;
  if (i >= K * N) return;
  int k = i / N, n = i - k * N;
  Wt[(size_t)n * K + k] = (bf16_t)W[i];
}

// Vproj f32 [T][NKV*HD] -> Vt bf16 [NKV][HD][T] (kv-contiguous rows for PV B-tiles)
__global__ void v_transpose_cvt(const float* __restrict__ Vp, bf16_t* __restrict__ Vt) {
  int i = blockIdx.x * blockDim.x + threadIdx.x;   // T*256
  int t = i >> 8;
  int c = i & 255;                                  // h*64 + d
  Vt[(size_t)c * T_SEQ + t] = (bf16_t)Vp[i];
}

// ---------------- WMMA GEMM with TDM-staged B: C[M][N] = A[M][K] * Bt[N][K]^T ----
// Block = 8 waves; all waves share n0 (64 cols). The 64x32 B slab for each K-step
// is DMA'd into LDS once per block by the Tensor Data Mover (double-buffered),
// then consumed by all 8 waves via ds_load_b128. Each wave computes a 32x64 tile.
// Requires M % 256 == 0, N % 64 == 0, K % 32 == 0 (true for all launches here).
__global__ __launch_bounds__(256) void gemm_bf16_f32_tdm(
    const bf16_t* __restrict__ A, const bf16_t* __restrict__ Bt,
    float* __restrict__ C, int M, int N, int K) {
  __shared__ __align__(16) bf16_t Bs[2][64][32];
  const int wave = threadIdx.x >> 5;
  const int lane = threadIdx.x & 31;
  const int m0 = (blockIdx.y * 8 + wave) * 32;
  const int n0 = blockIdx.x * 64;

  v8f acc[8] = {};

  // prologue: wave 0 kicks off the first B slab DMA
  if (wave == 0) {
    tdm_load_2d_bf16((unsigned)(uintptr_t)&Bs[0][0][0],
                     Bt + (size_t)n0 * K, 32, 64, (unsigned)K, (unsigned)N, (unsigned)K);
  }

  int buf = 0;
  for (int k = 0; k < K; k += 32, buf ^= 1) {
    if (wave == 0) {
      if (k + 32 < K) {
        tdm_load_2d_bf16((unsigned)(uintptr_t)&Bs[buf ^ 1][0][0],
                         Bt + (size_t)n0 * K + k + 32, 32, 64,
                         (unsigned)K, (unsigned)N, (unsigned)K);
        __builtin_amdgcn_s_wait_tensorcnt(1);   // current slab done (in-order TDM)
      } else {
        __builtin_amdgcn_s_wait_tensorcnt(0);
      }
    }
    __syncthreads();   // publish Bs[buf] to all waves

    v16bf a0 = load_tile16x32(A + (size_t)m0 * K + k, K);
    v16bf a1 = load_tile16x32(A + (size_t)(m0 + 16) * K + k, K);
#pragma unroll
    for (int j = 0; j < 4; ++j) {
      v16bf b = load_tile16x32(&Bs[buf][j * 16][0], 32);
      acc[j]     = __builtin_amdgcn_wmma_f32_16x16x32_bf16(false, a0, false, b, (short)0, acc[j],     false, false);
      acc[4 + j] = __builtin_amdgcn_wmma_f32_16x16x32_bf16(false, a1, false, b, (short)0, acc[4 + j], false, false);
    }
    __syncthreads();   // all waves done with Bs[buf] before TDM may overwrite it
  }

  // C/D layout: lanes 0-15 -> M = r, N = lane; lanes 16-31 -> M = 8+r, N = lane-16
  const int nn = lane & 15;
  const int mofs = (lane & 16) ? 8 : 0;
#pragma unroll
  for (int r = 0; r < 8; ++r) {
    size_t row0 = (size_t)(m0 + mofs + r) * N + n0 + nn;
    size_t row1 = (size_t)(m0 + 16 + mofs + r) * N + n0 + nn;
#pragma unroll
    for (int j = 0; j < 4; ++j) {
      C[row0 + j * 16] = acc[j][r];
      C[row1 + j * 16] = acc[4 + j][r];
    }
  }
}

// ---------------- RMSNorm + RoPE (Q & K fused), writes bf16 head-major -----------
__global__ __launch_bounds__(256) void qk_norm_rope(
    const float* __restrict__ Qp,   // [T][NH*HD]
    const float* __restrict__ Kp,   // [T][NKV*HD]
    const float* __restrict__ cosb, // [T][32]
    const float* __restrict__ sinb, // [T][32]
    const float* __restrict__ qw, const float* __restrict__ kw,
    bf16_t* __restrict__ Qn,        // [NH][T][HD]
    bf16_t* __restrict__ Kn) {      // [NKV][T][HD]
  const int wave = threadIdx.x >> 5;
  const int lane = threadIdx.x & 31;
  const int row = blockIdx.x * 8 + wave;          // T*(NH+NKV) rows
  const int t = row / (NH + NKV);
  const int hh = row - t * (NH + NKV);
  const bool isq = hh < NH;
  const float* src = isq ? (Qp + (size_t)t * (NH * HD) + hh * HD)
                         : (Kp + (size_t)t * (NKV * HD) + (hh - NH) * HD);
  const float* w = isq ? qw : kw;

  float x1 = src[lane];
  float x2 = src[lane + 32];
  float ss = x1 * x1 + x2 * x2;
#pragma unroll
  for (int m = 1; m < 32; m <<= 1) ss += __shfl_xor(ss, m, 32);
  float inv = rsqrtf(ss * (1.0f / HD) + RMS_EPS);
  x1 = x1 * inv * w[lane];
  x2 = x2 * inv * w[lane + 32];

  float c = cosb[(size_t)t * 32 + lane];
  float s = sinb[(size_t)t * 32 + lane];
  float o1 = x1 * c - x2 * s;
  float o2 = x1 * s + x2 * c;
  float sc = isq ? ATTN_SCALE : 1.0f;  // fold 1/sqrt(D) into Q

  bf16_t* dst = isq ? (Qn + ((size_t)hh * T_SEQ + t) * HD)
                    : (Kn + ((size_t)(hh - NH) * T_SEQ + t) * HD);
  dst[lane]      = (bf16_t)(o1 * sc);
  dst[lane + 32] = (bf16_t)(o2 * sc);
}

// ---------------- Flash attention (causal, GQA) ----------------------------------
// One wave per (head, 16-query tile). kv processed in 32-wide blocks:
// 4 WMMAs for scores (K = D = 64), online softmax, P via LDS, 4 WMMAs for PV.
__global__ __launch_bounds__(256) void flash_attn(
    const bf16_t* __restrict__ Qn,  // [NH][T][HD], scale pre-folded
    const bf16_t* __restrict__ Kn,  // [NKV][T][HD]
    const bf16_t* __restrict__ Vt,  // [NKV][HD][T]
    bf16_t* __restrict__ AO) {      // [T][NH*HD]
  __shared__ __align__(16) bf16_t Pbuf[8][16][32];
  const int wave = threadIdx.x >> 5;
  const int lane = threadIdx.x & 31;
  const int gw = blockIdx.x * 8 + wave;      // 0 .. T/16*NH-1
  const int h  = gw & (NH - 1);
  const int q0 = (gw >> 4) * 16;
  const int hkv = h >> 2;                    // G = NH/NKV = 4
  const int nn = lane & 15;
  const int mofs = (lane & 16) ? 8 : 0;

  const bf16_t* Qh = Qn + ((size_t)h * T_SEQ + q0) * HD;
  const bf16_t* Kh = Kn + (size_t)hkv * T_SEQ * HD;
  const bf16_t* Vh = Vt + (size_t)hkv * HD * T_SEQ;

  v16bf qa0 = load_tile16x32(Qh, HD);        // d = 0..31
  v16bf qa1 = load_tile16x32(Qh + 32, HD);   // d = 32..63

  v8f o0 = {}, o1 = {}, o2 = {}, o3 = {};
  float mst[8], lst[8];
#pragma unroll
  for (int r = 0; r < 8; ++r) { mst[r] = NEG_INF; lst[r] = 0.f; }

  for (int kb = 0; kb < q0 + 16; kb += 32) {
    // ---- scores: two 16x16 tiles covering kv [kb, kb+32) ----
    v8f s0 = {}, s1 = {};
    {
      v16bf b = load_tile16x32(Kh + (size_t)kb * HD, HD);
      s0 = __builtin_amdgcn_wmma_f32_16x16x32_bf16(false, qa0, false, b, (short)0, s0, false, false);
      b = load_tile16x32(Kh + (size_t)kb * HD + 32, HD);
      s0 = __builtin_amdgcn_wmma_f32_16x16x32_bf16(false, qa1, false, b, (short)0, s0, false, false);
      b = load_tile16x32(Kh + (size_t)(kb + 16) * HD, HD);
      s1 = __builtin_amdgcn_wmma_f32_16x16x32_bf16(false, qa0, false, b, (short)0, s1, false, false);
      b = load_tile16x32(Kh + (size_t)(kb + 16) * HD + 32, HD);
      s1 = __builtin_amdgcn_wmma_f32_16x16x32_bf16(false, qa1, false, b, (short)0, s1, false, false);
    }
    // ---- causal mask in C layout ----
#pragma unroll
    for (int r = 0; r < 8; ++r) {
      const int tt = q0 + mofs + r;
      if (kb + nn > tt)      s0[r] = NEG_INF;
      if (kb + 16 + nn > tt) s1[r] = NEG_INF;
    }
    // ---- online softmax update + P -> LDS ----
#pragma unroll
    for (int r = 0; r < 8; ++r) {
      float bm = redmax16(fmaxf(s0[r], s1[r]));
      float mnew = fmaxf(mst[r], bm);
      float alpha = __expf(mst[r] - mnew);
      mst[r] = mnew;
      float p0 = __expf(s0[r] - mnew);
      float p1 = __expf(s1[r] - mnew);
      float rs = redsum16(p0 + p1);
      lst[r] = lst[r] * alpha + rs;
      o0[r] *= alpha; o1[r] *= alpha; o2[r] *= alpha; o3[r] *= alpha;
      Pbuf[wave][mofs + r][nn]      = (bf16_t)p0;
      Pbuf[wave][mofs + r][16 + nn] = (bf16_t)p1;
    }
    asm volatile("s_wait_dscnt 0" ::: "memory");  // LDS stores visible to A-layout reload
    // ---- PV: A = P (16x32), B = V^T rows (d-contiguous kv) ----
    v16bf pa = load_tile16x32(&Pbuf[wave][0][0], 32);
    {
      v16bf bv = load_tile16x32(Vh + (size_t)0  * T_SEQ + kb, T_SEQ);
      o0 = __builtin_amdgcn_wmma_f32_16x16x32_bf16(false, pa, false, bv, (short)0, o0, false, false);
      bv = load_tile16x32(Vh + (size_t)16 * T_SEQ + kb, T_SEQ);
      o1 = __builtin_amdgcn_wmma_f32_16x16x32_bf16(false, pa, false, bv, (short)0, o1, false, false);
      bv = load_tile16x32(Vh + (size_t)32 * T_SEQ + kb, T_SEQ);
      o2 = __builtin_amdgcn_wmma_f32_16x16x32_bf16(false, pa, false, bv, (short)0, o2, false, false);
      bv = load_tile16x32(Vh + (size_t)48 * T_SEQ + kb, T_SEQ);
      o3 = __builtin_amdgcn_wmma_f32_16x16x32_bf16(false, pa, false, bv, (short)0, o3, false, false);
    }
  }

  // ---- epilogue: normalize and write bf16 attn output [T][NH*HD] ----
#pragma unroll
  for (int r = 0; r < 8; ++r) {
    float inv = 1.0f / lst[r];
    size_t row = (size_t)(q0 + mofs + r) * (NH * HD) + h * HD + nn;
    AO[row +  0] = (bf16_t)(o0[r] * inv);
    AO[row + 16] = (bf16_t)(o1[r] * inv);
    AO[row + 32] = (bf16_t)(o2[r] * inv);
    AO[row + 48] = (bf16_t)(o3[r] * inv);
  }
}

// ---------------- host-side orchestration ----------------
extern "C" void kernel_launch(void* const* d_in, const int* in_sizes, int n_in,
                              void* d_out, int out_size, void* d_ws, size_t ws_size,
                              hipStream_t stream) {
  (void)in_sizes; (void)n_in; (void)out_size; (void)ws_size;
  const float* hidden = (const float*)d_in[0];
  const float* cosb   = (const float*)d_in[1];
  const float* sinb   = (const float*)d_in[2];
  const float* Wq     = (const float*)d_in[3];
  const float* Wk     = (const float*)d_in[4];
  const float* Wv     = (const float*)d_in[5];
  const float* Wo     = (const float*)d_in[6];
  const float* qw     = (const float*)d_in[7];
  const float* kw     = (const float*)d_in[8];
  float* out = (float*)d_out;

  // workspace carve-up (256B aligned)
  char* ws = (char*)d_ws;
  size_t off = 0;
  auto carve = [&](size_t bytes) -> void* {
    void* p = ws + off;
    off += (bytes + 255) & ~(size_t)255;
    return p;
  };
  bf16_t* Xb   = (bf16_t*)carve((size_t)T_SEQ * HID * 2);
  bf16_t* Wqt  = (bf16_t*)carve((size_t)HID * (NH * HD) * 2);
  bf16_t* Wkt  = (bf16_t*)carve((size_t)HID * (NKV * HD) * 2);
  bf16_t* Wvt  = (bf16_t*)carve((size_t)HID * (NKV * HD) * 2);
  bf16_t* Wot  = (bf16_t*)carve((size_t)(NH * HD) * HID * 2);
  float*  Qp   = (float*) carve((size_t)T_SEQ * (NH * HD) * 4);
  float*  Kp   = (float*) carve((size_t)T_SEQ * (NKV * HD) * 4);
  float*  Vp   = (float*) carve((size_t)T_SEQ * (NKV * HD) * 4);
  bf16_t* Qnb  = (bf16_t*)carve((size_t)NH  * T_SEQ * HD * 2);
  bf16_t* Knb  = (bf16_t*)carve((size_t)NKV * T_SEQ * HD * 2);
  bf16_t* Vtb  = (bf16_t*)carve((size_t)NKV * HD * T_SEQ * 2);
  bf16_t* AOb  = (bf16_t*)carve((size_t)T_SEQ * (NH * HD) * 2);

  const int BLK = 256;

  // 1) conversions
  cvt_f32_bf16<<<(T_SEQ * HID) / BLK, BLK, 0, stream>>>(hidden, Xb, T_SEQ * HID);
  transpose_cvt<<<(HID * NH * HD) / BLK, BLK, 0, stream>>>(Wq, Wqt, HID, NH * HD);
  transpose_cvt<<<(HID * NKV * HD) / BLK, BLK, 0, stream>>>(Wk, Wkt, HID, NKV * HD);
  transpose_cvt<<<(HID * NKV * HD) / BLK, BLK, 0, stream>>>(Wv, Wvt, HID, NKV * HD);
  transpose_cvt<<<(NH * HD * HID) / BLK, BLK, 0, stream>>>(Wo, Wot, NH * HD, HID);

  // 2) QKV projections (WMMA + TDM-staged B)
  gemm_bf16_f32_tdm<<<dim3((NH * HD) / 64, T_SEQ / 256), BLK, 0, stream>>>(Xb, Wqt, Qp, T_SEQ, NH * HD, HID);
  gemm_bf16_f32_tdm<<<dim3((NKV * HD) / 64, T_SEQ / 256), BLK, 0, stream>>>(Xb, Wkt, Kp, T_SEQ, NKV * HD, HID);
  gemm_bf16_f32_tdm<<<dim3((NKV * HD) / 64, T_SEQ / 256), BLK, 0, stream>>>(Xb, Wvt, Vp, T_SEQ, NKV * HD, HID);

  // 3) RMSNorm + RoPE (Q,K) and V transpose
  qk_norm_rope<<<(T_SEQ * (NH + NKV)) / 8, BLK, 0, stream>>>(Qp, Kp, cosb, sinb, qw, kw, Qnb, Knb);
  v_transpose_cvt<<<(T_SEQ * NKV * HD) / BLK, BLK, 0, stream>>>(Vp, Vtb);

  // 4) causal flash attention (WMMA)
  flash_attn<<<(T_SEQ / 16 * NH) / 8, BLK, 0, stream>>>(Qnb, Knb, Vtb, AOb);

  // 5) output projection (WMMA + TDM-staged B) -> f32 d_out
  gemm_bf16_f32_tdm<<<dim3(HID / 64, T_SEQ / 256), BLK, 0, stream>>>(AOb, Wot, out, T_SEQ, HID, NH * HD);
}
